// CHESHIRE_90374701843019
// MI455X (gfx1250) — matrix operand order repaired and checked by reading, not compiled, not run on hardware
//
#include <hip/hip_runtime.h>

// ---- problem constants (match reference) ----
#define NN    2000
#define FF    256
#define EMB   128
#define CONV  128
#define EE    20000
#define SS    8

typedef _Float16 h4  __attribute__((ext_vector_type(4)));
typedef _Float16 h8  __attribute__((ext_vector_type(8)));
typedef _Float16 h16 __attribute__((ext_vector_type(16)));
typedef float    f4  __attribute__((ext_vector_type(4)));
typedef float    v8f __attribute__((ext_vector_type(8)));

static __device__ __forceinline__ v8f wmma_f16(h16 a, h16 b, v8f c) {
  // D = A(16x32 f16) x B(32x16 f16) + C(16x16 f32)
  return __builtin_amdgcn_wmma_f32_16x16x32_f16(false, a, false, b, (short)0, c,
                                                false, false);
}

static __device__ __forceinline__ h16 cat8(h8 lo, h8 hi) {
  return __builtin_shufflevector(lo, hi, 0, 1, 2, 3, 4, 5, 6, 7,
                                 8, 9, 10, 11, 12, 13, 14, 15);
}

// Async global -> LDS copy, 16 bytes per lane. Tracked by ASYNCcnt.
// (inline asm: portable across ROCm7.2 / amdgpu-toolchain builtin arity skew)
static __device__ __forceinline__ void async_g2l_b128(unsigned lds_off,
                                                      unsigned g_off,
                                                      const void* g_base) {
  asm volatile("global_load_async_to_lds_b128 %0, %1, %2"
               :
               : "v"(lds_off), "v"(g_off), "s"(g_base)
               : "memory");
}

static __device__ __forceinline__ void wait_asynccnt0() {
  asm volatile("s_wait_asynccnt 0x0" ::: "memory");
}

// ---------------------------------------------------------------------------
// Kernel 1: weight prep.
//   WencT[n][k] = W_enc[k][n]            (f16, K-contiguous for B-fragments)
//   WaT[n][k]   = (W0 + W1/7 + (2/49-1)W2)[k][n]   (f16)
//   WbT[n][k]   = (-W1/7 + (12/49)W2)[k][n]        (f16)
// ---------------------------------------------------------------------------
__global__ void che_prep(const float* __restrict__ Wenc,
                         const float* __restrict__ chebW,
                         _Float16* __restrict__ WencT,
                         _Float16* __restrict__ WaT,
                         _Float16* __restrict__ WbT) {
  int idx = blockIdx.x * 256 + threadIdx.x;
  if (idx < EMB * FF) {                       // 32768: encoder weights
    int n = idx >> 8, k = idx & 255;
    WencT[idx] = (_Float16)Wenc[k * EMB + n];
  } else if (idx < EMB * FF + EMB * CONV) {   // 16384: cheb combos
    int j = idx - EMB * FF;
    int n = j & 127, k = j >> 7;              // j = k*128 + n
    float w0 = chebW[0 * EMB * CONV + k * CONV + n];
    float w1 = chebW[1 * EMB * CONV + k * CONV + n];
    float w2 = chebW[2 * EMB * CONV + k * CONV + n];
    WaT[n * EMB + k] = (_Float16)(w0 + w1 * (1.0f / 7.0f) + w2 * (2.0f / 49.0f - 1.0f));
    WbT[n * EMB + k] = (_Float16)(w2 * (12.0f / 49.0f) - w1 * (1.0f / 7.0f));
  }
}

// ---------------------------------------------------------------------------
// Kernel 2: encoder  enc = clip(feature @ W_enc + b_enc, -1, 1)   [2000,128]
// One wave per 16-row tile (125 tiles). WMMA f16, K=256 in 8 chunks of 32.
// feature (2MB) and WencT (64KB) are L2-resident; run only 125 waves total.
// ---------------------------------------------------------------------------
__global__ __launch_bounds__(128) void che_enc(const float* __restrict__ feature,
                                               const _Float16* __restrict__ WencT,
                                               const float* __restrict__ benc,
                                               float* __restrict__ enc) {
  const int wv = threadIdx.x >> 5;
  const int t  = blockIdx.x * 4 + wv;          // row tile id
  if (t >= NN / 16) return;                    // uniform per wave
  const int l = threadIdx.x & 31;
  const int m = l & 15;                        // row within tile / column lane
  const int h = l >> 4;                        // wave half

  v8f acc[8] = {};
#pragma unroll
  for (int kk = 0; kk < FF / 32; ++kk) {
    // A fragment: row m, K = kk*32 + h*8 + {0..7} and +16..23 (f32 -> f16)
    const float* fp = feature + (t * 16 + m) * FF + kk * 32 + h * 8;
    f4 q0 = *(const f4*)(fp + 0);
    f4 q1 = *(const f4*)(fp + 4);
    f4 q2 = *(const f4*)(fp + 16);
    f4 q3 = *(const f4*)(fp + 20);
    h8 lo = __builtin_shufflevector(__builtin_convertvector(q0, h4),
                                    __builtin_convertvector(q1, h4),
                                    0, 1, 2, 3, 4, 5, 6, 7);
    h8 hi = __builtin_shufflevector(__builtin_convertvector(q2, h4),
                                    __builtin_convertvector(q3, h4),
                                    0, 1, 2, 3, 4, 5, 6, 7);
    h16 a = cat8(lo, hi);
#pragma unroll
    for (int n = 0; n < 8; ++n) {
      // B fragment: col N = n*16+m, K = kk*32 + h*8 + {0..7} and +16..23
      const _Float16* bp = WencT + (n * 16 + m) * FF + kk * 32 + h * 8;
      h16 b = cat8(*(const h8*)bp, *(const h8*)(bp + 16));
      acc[n] = wmma_f16(a, b, acc[n]);
    }
  }
  // C layout: lane half h -> rows v + 8h ; col N = n*16 + m
#pragma unroll
  for (int n = 0; n < 8; ++n) {
    float be = benc[n * 16 + m];
#pragma unroll
    for (int v = 0; v < 8; ++v) {
      float x = fminf(fmaxf(acc[n][v] + be, -1.0f), 1.0f);
      enc[(t * 16 + v + 8 * h) * EMB + n * 16 + m] = x;
    }
  }
}

// ---------------------------------------------------------------------------
// Kernel 3: fused gather + GraphNorm + Cheb(GEMM) + pooling + head.
// One wave handles 2 hyperedges (= 16 expanded rows, one WMMA M-tile).
//   out_i = Tx0_i @ Wa + s_e @ Wb ; clip ; per-hyperedge (max-min, rms) pools;
//   sigmoid(y @ W_out + b_out)
// Wa is async-copied to LDS at block start (overlapped with the gather /
// GraphNorm phase) so WMMA B-fragments come from conflict-free ds_load_b128.
// ---------------------------------------------------------------------------
__global__ __launch_bounds__(128) void che_main(const float* __restrict__ enc,
                                                const _Float16* __restrict__ WaT,
                                                const _Float16* __restrict__ WbT,
                                                const float* __restrict__ gnw,
                                                const float* __restrict__ gnb,
                                                const float* __restrict__ gms,
                                                const float* __restrict__ chebb,
                                                const float* __restrict__ Wout,
                                                const float* __restrict__ bout,
                                                const int* __restrict__ hnodes,
                                                float* __restrict__ out) {
  __shared__ __align__(32) _Float16 sWa[EMB * EMB];     // 32KB, block-shared
  __shared__ __align__(32) _Float16 sTx0[4][16 * EMB];  // per-wave 16x128 f16 tile
  __shared__ __align__(32) _Float16 sSh[4][2 * EMB];    // per-wave hyperedge sums

  const int wv = threadIdx.x >> 5;
  const int l  = threadIdx.x & 31;
  const int m  = l & 15;
  const int h  = l >> 4;
  const int wt = blockIdx.x * 4 + wv;   // wave task: 2500 blocks * 4 waves = 10000
  const int e0 = wt * 2;                // hyperedges e0, e0+1

  // ---- kick off async Wa copy: 32KB = 2048 x b128, 16 per thread ----
  {
    const unsigned base = (unsigned)(uintptr_t)(&sWa[0]);
#pragma unroll
    for (int it = 0; it < 16; ++it) {
      unsigned off = (it * 128 + threadIdx.x) * 16;
      async_g2l_b128(base + off, off, (const void*)WaT);
    }
  }

  // ---- phase 1: gather + GraphNorm, lane owns channels 4l..4l+3 ----
  f4 gms4 = *(const f4*)(gms + 4 * l);
  f4 gnw4 = *(const f4*)(gnw + 4 * l);
  f4 gnb4 = *(const f4*)(gnb + 4 * l);
  f4 xv[16];
#pragma unroll
  for (int r = 0; r < 16; ++r) {
    int node = hnodes[(e0 + (r >> 3)) * SS + (r & 7)];
    xv[r] = *(const f4*)(enc + node * EMB + 4 * l);
  }
#pragma unroll
  for (int e = 0; e < 2; ++e) {
    f4 sum = {0.0f, 0.0f, 0.0f, 0.0f};
#pragma unroll
    for (int r = 0; r < 8; ++r) sum += xv[e * 8 + r];
    f4 mg = sum * 0.125f * gms4;                       // mean * gn_mean_scale
    f4 var = {0.0f, 0.0f, 0.0f, 0.0f};
#pragma unroll
    for (int r = 0; r < 8; ++r) { f4 d = xv[e * 8 + r] - mg; var += d * d; }
    var *= 0.125f;
    f4 scale;
#pragma unroll
    for (int j = 0; j < 4; ++j) scale[j] = rsqrtf(var[j] + 1e-5f);
    scale *= gnw4;
    f4 sv = {0.0f, 0.0f, 0.0f, 0.0f};
#pragma unroll
    for (int r = 0; r < 8; ++r) {
      f4 t = (xv[e * 8 + r] - mg) * scale + gnb4;      // Tx0 row
      sv += t;
      *(h4*)(&sTx0[wv][(e * 8 + r) * EMB + 4 * l]) = __builtin_convertvector(t, h4);
    }
    *(h4*)(&sSh[wv][e * EMB + 4 * l]) = __builtin_convertvector(sv, h4);
  }
  wait_asynccnt0();      // Wa copy landed (overlapped with all of phase 1)
  __syncthreads();

  // ---- phase 2: D = Tx0 @ Wa + Srep @ Wb via WMMA, K=128 in 4 chunks ----
  v8f acc[8] = {};
#pragma unroll
  for (int kk = 0; kk < 4; ++kk) {
    // A fragment from row-major LDS tile: row m, K = kk*32 + h*8 {0..7,+16..23}
    const _Float16* ap = &sTx0[wv][m * EMB + kk * 32 + h * 8];
    h16 a = cat8(*(const h8*)ap, *(const h8*)(ap + 16));
    // A2 fragment: rows 0-7 replicate s_e0, rows 8-15 replicate s_e1
    const _Float16* sp = &sSh[wv][(m >> 3) * EMB + kk * 32 + h * 8];
    h16 a2 = cat8(*(const h8*)sp, *(const h8*)(sp + 16));
#pragma unroll
    for (int n = 0; n < 8; ++n) {
      int N  = n * 16 + m;
      h16 ba = cat8(*(const h8*)(&sWa[N * EMB + kk * 32 + h * 8]), *(const h8*)(&sWa[N * EMB + kk * 32 + h * 8 + 16]));   // LDS
      h16 bb = cat8(*(const h8*)(WbT + N * EMB + kk * 32 + h * 8), *(const h8*)(WbT + N * EMB + kk * 32 + h * 8 + 16));   // L2
      acc[n] = wmma_f16(a, ba, acc[n]);
      acc[n] = wmma_f16(a2, bb, acc[n]);
    }
  }

  // ---- phase 3: bias + clip + per-hyperedge pools + output head ----
  // C layout: half h owns hyperedge e0+h exclusively (rows v or v+8).
  float partial = 0.0f;
#pragma unroll
  for (int n = 0; n < 8; ++n) {
    int N = n * 16 + m;
    float bias = chebb[N];
    float mx = -1e30f, mn = 1e30f, ss = 0.0f;
#pragma unroll
    for (int v = 0; v < 8; ++v) {
      float x = fminf(fmaxf(acc[n][v] + bias, -1.0f), 1.0f);
      mx = fmaxf(mx, x);
      mn = fminf(mn, x);
      ss += x * x;
    }
    float nrm = sqrtf(ss * 0.125f);
    partial += (mx - mn) * Wout[N] + nrm * Wout[CONV + N];
  }
  // reduce over the 16 lanes of this wave half
#pragma unroll
  for (int off = 8; off >= 1; off >>= 1) partial += __shfl_xor(partial, off, 32);
  if (m == 0) {
    float z = partial + bout[0];
    out[e0 + h] = 1.0f / (1.0f + __expf(-z));
  }
}

// ---------------------------------------------------------------------------
// Host launcher
// ---------------------------------------------------------------------------
extern "C" void kernel_launch(void* const* d_in, const int* in_sizes, int n_in,
                              void* d_out, int out_size, void* d_ws, size_t ws_size,
                              hipStream_t stream) {
  const float* feature = (const float*)d_in[0];   // [2000,256]
  const float* W_enc   = (const float*)d_in[1];   // [256,128]
  const float* b_enc   = (const float*)d_in[2];   // [128]
  const float* gn_w    = (const float*)d_in[3];   // [128]
  const float* gn_b    = (const float*)d_in[4];   // [128]
  const float* gn_ms   = (const float*)d_in[5];   // [128]
  const float* cheb_W  = (const float*)d_in[6];   // [3,128,128]
  const float* cheb_b  = (const float*)d_in[7];   // [128]
  const float* W_out   = (const float*)d_in[8];   // [256,1]
  const float* b_out   = (const float*)d_in[9];   // [1]
  const int*   hnodes  = (const int*)d_in[10];    // [20000,8]
  // d_in[11], d_in[12] (row/col) unused: clique structure folded analytically.
  float* out = (float*)d_out;                     // [20000]

  char* ws = (char*)d_ws;
  float*     enc   = (float*)ws;                               // 1,024,000 B
  _Float16*  WencT = (_Float16*)(ws + 1024000);                //    65,536 B
  _Float16*  WaT   = (_Float16*)(ws + 1024000 + 65536);        //    32,768 B
  _Float16*  WbT   = (_Float16*)(ws + 1024000 + 65536 + 32768);//    32,768 B

  che_prep<<<192, 256, 0, stream>>>(W_enc, cheb_W, WencT, WaT, WbT);
  che_enc<<<32, 128, 0, stream>>>(feature, WencT, b_enc, enc);
  che_main<<<2500, 128, 0, stream>>>(enc, WaT, WbT, gn_w, gn_b, gn_ms,
                                     cheb_b, W_out, b_out, hnodes, out);
}
